// GAT_11802570130081
// MI455X (gfx1250) — compile-verified, matrix operand order checked by reading, not run on hardware
//
#include <hip/hip_runtime.h>
#include <hip/hip_bf16.h>
#include <math.h>

// ---------------- problem constants (from reference) ----------------
#define NN      20000        // nodes
#define NE      320000       // edges
#define NG      128          // graphs
#define DIN     92
#define KPAD    96           // 92 padded to 3 x k32 WMMA steps
#define DH      91
#define HEADS   8
#define H1      728          // HEADS*DH
#define NCAT    1456         // W_l || W_r  (2*H1)
#define NTILES  91           // 1456/16
#define MTILES  1250         // 20000/16
#define NEG_SLOPE 0.2f

typedef __attribute__((ext_vector_type(16))) _Float16 v16h;
typedef __attribute__((ext_vector_type(8)))  _Float16 v8h;
typedef __attribute__((ext_vector_type(8)))  float    v8f;

// order-preserving float<->uint encoding for atomicMax-based segment max
__device__ __forceinline__ unsigned encf(float f) {
    int i = __float_as_int(f);
    return (i >= 0) ? ((unsigned)i ^ 0x80000000u) : ~(unsigned)i;
}
__device__ __forceinline__ float decf(unsigned u) {
    return (u >= 0x80000000u) ? __int_as_float((int)(u ^ 0x80000000u))
                              : __int_as_float((int)~u);
}
__device__ __forceinline__ float lrelu(float v) { return v > 0.f ? v : NEG_SLOPE * v; }

// ---------------- utility ----------------
__global__ void zero32(unsigned* __restrict__ p, long long n) {
    long long i = (long long)blockIdx.x * blockDim.x + threadIdx.x;
    long long stride = (long long)gridDim.x * blockDim.x;
    for (; i < n; i += stride) p[i] = 0u;
}

// x[20000,92] f32 -> Xh[20000,96] f16 (zero-padded K)
__global__ void pack_x(const float* __restrict__ x, _Float16* __restrict__ Xh) {
    int i = blockIdx.x * blockDim.x + threadIdx.x;
    if (i >= NN * KPAD) return;
    int n = i / KPAD, k = i % KPAD;
    Xh[i] = (k < DIN) ? (_Float16)x[n * DIN + k] : (_Float16)0.f;
}

// Pack W_l||W_r into per-lane WMMA B-fragment order:
// Wpack[nt][s][lane][j], j<8: K=32s+(lane<16?0:8)+j ; j>=8: K=32s+(lane<16?16:24)+(j-8)
// column n = nt*16 + (lane&15).  Also build biascat[1456].
__global__ void pack_w(const float* __restrict__ Wl, const float* __restrict__ Wr,
                       const float* __restrict__ bl, const float* __restrict__ br,
                       _Float16* __restrict__ Wpack, float* __restrict__ biascat) {
    int i = blockIdx.x * blockDim.x + threadIdx.x;
    const int total = NTILES * 3 * 32 * 16;
    if (i < NCAT) biascat[i] = (i < H1) ? bl[i] : br[i - H1];
    if (i >= total) return;
    int j = i & 15;
    int lane = (i >> 4) & 31;
    int rest = i >> 9;
    int s = rest % 3;
    int nt = rest / 3;
    int n = nt * 16 + (lane & 15);
    int k = (j < 8) ? (32 * s + ((lane < 16) ? 0 : 8) + j)
                    : (32 * s + ((lane < 16) ? 16 : 24) + (j - 8));
    float v = 0.f;
    if (k < DIN) v = (n < H1) ? Wl[k * H1 + n] : Wr[k * H1 + (n - H1)];
    Wpack[i] = (_Float16)v;
}

// ---------------- WMMA GEMM: XLR[20000,1456] = Xh @ Wcat + biascat ----------------
// 8 waves / block; each wave owns one 16x16 output tile. 3 x v_wmma_f32_16x16x32_f16.
__global__ __launch_bounds__(256) void gemm_wmma(
    const _Float16* __restrict__ Xh, const _Float16* __restrict__ Wpack,
    const float* __restrict__ biascat, float* __restrict__ XLR) {
    const int lane  = threadIdx.x & 31;
    const int wave  = threadIdx.x >> 5;
    const int mtile = blockIdx.x * 8 + wave;
    const int ntile = blockIdx.y;
    if (mtile >= MTILES) return;          // wave-uniform: EXEC stays all-ones for WMMA

    const int row = mtile * 16 + (lane & 15);
    const int hi  = lane >> 4;            // 0: lanes 0-15, 1: lanes 16-31
    const _Float16* abase = Xh + (size_t)row * KPAD;
    const _Float16* bbase = Wpack + (((size_t)ntile * 3) * 32 + lane) * 16;

    v8f acc = {};
#pragma unroll
    for (int s = 0; s < 3; ++s) {
        // A fragment: ISA 16-bit 16x32 layout, two contiguous 16B chunks per lane
        v8h a0 = *(const v8h*)(abase + 32 * s + hi * 8);        // K 0-7 / 8-15
        v8h a1 = *(const v8h*)(abase + 32 * s + 16 + hi * 8);   // K 16-23 / 24-31
        v16h a;
#pragma unroll
        for (int j = 0; j < 8; ++j) { a[j] = a0[j]; a[8 + j] = a1[j]; }
        // B fragment: pre-packed contiguous 32B per lane
        v16h b = *(const v16h*)(bbase + (size_t)s * 512);
        acc = __builtin_amdgcn_wmma_f32_16x16x32_f16(
            /*neg_a=*/false, a, /*neg_b=*/false, b,
            /*c_mod=*/(short)0, acc, /*reuse_a=*/false, /*reuse_b=*/false);
    }
    // C/D layout: lane&15 = N column; lanes 0-15 rows 0-7, lanes 16-31 rows 8-15
    const int ncol = ntile * 16 + (lane & 15);
    const float bv = biascat[ncol];
    float* out = XLR + (size_t)(mtile * 16 + hi * 8) * NCAT + ncol;
#pragma unroll
    for (int v = 0; v < 8; ++v) out[(size_t)v * NCAT] = acc[v] + bv;
}

// ---------------- layer-1 edge stages ----------------
// logits[e,h] = sum_c att1[h,c]*lrelu(xl[src,h,c]+xr[dst,h,c]); segment max via atomicMax
__global__ void edge_logits1(const int* __restrict__ src, const int* __restrict__ dst,
                             const float* __restrict__ XLR, const float* __restrict__ att1,
                             float* __restrict__ logits, unsigned* __restrict__ maxenc) {
    int t = blockIdx.x * blockDim.x + threadIdx.x;
    if (t >= NE * HEADS) return;
    int e = t >> 3, h = t & 7;
    int s = src[e], d = dst[e];
    const float* xl = XLR + (size_t)s * NCAT + h * DH;
    const float* xr = XLR + (size_t)d * NCAT + H1 + h * DH;
    const float* at = att1 + h * DH;
    float acc = 0.f;
    for (int c = 0; c < DH; ++c) acc += at[c] * lrelu(xl[c] + xr[c]);
    logits[t] = acc;
    atomicMax(&maxenc[d * HEADS + h], encf(acc));
}

__global__ void edge_alpha1(const int* __restrict__ dst, float* __restrict__ logits,
                            const unsigned* __restrict__ maxenc, float* __restrict__ denom) {
    int t = blockIdx.x * blockDim.x + threadIdx.x;
    if (t >= NE * HEADS) return;
    int e = t >> 3, h = t & 7;
    int d = dst[e];
    float a = __expf(logits[t] - decf(maxenc[d * HEADS + h]));
    logits[t] = a;                                   // reuse as alpha
    atomicAdd(&denom[d * HEADS + h], a);
}

// out1[dst, c] += alpha/(denom+eps) * xl[src, c]   (one thread per edge-channel)
__global__ void edge_message1(const int* __restrict__ src, const int* __restrict__ dst,
                              const float* __restrict__ XLR, const float* __restrict__ alpha,
                              const float* __restrict__ denom, float* __restrict__ out1) {
    long long t = (long long)blockIdx.x * blockDim.x + threadIdx.x;
    if (t >= (long long)NE * H1) return;
    int e = (int)(t / H1);
    int c = (int)(t % H1);
    int h = c / DH;
    int s = src[e], d = dst[e];
    float w = alpha[e * HEADS + h] / (denom[d * HEADS + h] + 1e-16f);
    atomicAdd(&out1[(size_t)d * H1 + c], w * XLR[(size_t)s * NCAT + c]);
}

__global__ void bias_elu1(float* __restrict__ out1, const float* __restrict__ bias1) {
    long long t = (long long)blockIdx.x * blockDim.x + threadIdx.x;
    if (t >= (long long)NN * H1) return;
    int c = (int)(t % H1) + (int)((t / H1) % 1) * 0; // channel within H1
    c = (int)(t % H1);
    int h = (int)((t % H1)); (void)h;
    float v = out1[t] + bias1[t % H1];
    out1[t] = v > 0.f ? v : (__expf(v) - 1.f);
}

// ---------------- layer-2 (728 -> 1, heads=1) ----------------
// wave-per-node GEMV with shuffle reduction
__global__ __launch_bounds__(256) void gemv2(const float* __restrict__ h1,
                                             const float* __restrict__ Wl2, const float* __restrict__ bl2,
                                             const float* __restrict__ Wr2, const float* __restrict__ br2,
                                             float* __restrict__ hl2, float* __restrict__ hr2) {
    int lane = threadIdx.x & 31;
    int node = blockIdx.x * 8 + (threadIdx.x >> 5);
    if (node >= NN) return;
    const float* hp = h1 + (size_t)node * H1;
    float al = 0.f, ar = 0.f;
    for (int c = lane; c < H1; c += 32) {
        float v = hp[c];
        al += v * Wl2[c];
        ar += v * Wr2[c];
    }
    for (int off = 16; off; off >>= 1) {
        al += __shfl_xor(al, off, 32);
        ar += __shfl_xor(ar, off, 32);
    }
    if (lane == 0) { hl2[node] = al + bl2[0]; hr2[node] = ar + br2[0]; }
}

__global__ void edge_logits2(const int* __restrict__ src, const int* __restrict__ dst,
                             const float* __restrict__ hl2, const float* __restrict__ hr2,
                             const float* __restrict__ att2,
                             float* __restrict__ logits2, unsigned* __restrict__ max2) {
    int e = blockIdx.x * blockDim.x + threadIdx.x;
    if (e >= NE) return;
    int d = dst[e];
    float l = att2[0] * lrelu(hl2[src[e]] + hr2[d]);
    logits2[e] = l;
    atomicMax(&max2[d], encf(l));
}

__global__ void edge_alpha2(const int* __restrict__ dst, float* __restrict__ logits2,
                            const unsigned* __restrict__ max2, float* __restrict__ denom2) {
    int e = blockIdx.x * blockDim.x + threadIdx.x;
    if (e >= NE) return;
    int d = dst[e];
    float a = __expf(logits2[e] - decf(max2[d]));
    logits2[e] = a;
    atomicAdd(&denom2[d], a);
}

__global__ void edge_message2(const int* __restrict__ src, const int* __restrict__ dst,
                              const float* __restrict__ hl2, const float* __restrict__ alpha2,
                              const float* __restrict__ denom2, float* __restrict__ out2) {
    int e = blockIdx.x * blockDim.x + threadIdx.x;
    if (e >= NE) return;
    int d = dst[e];
    atomicAdd(&out2[d], alpha2[e] / (denom2[d] + 1e-16f) * hl2[src[e]]);
}

__global__ void pool(const float* __restrict__ out2, const float* __restrict__ bias2,
                     const int* __restrict__ batch, float* __restrict__ dout) {
    int n = blockIdx.x * blockDim.x + threadIdx.x;
    if (n >= NN) return;
    atomicAdd(&dout[batch[n]], out2[n] + bias2[0]);
}

// ---------------- launch ----------------
extern "C" void kernel_launch(void* const* d_in, const int* in_sizes, int n_in,
                              void* d_out, int out_size, void* d_ws, size_t ws_size,
                              hipStream_t stream) {
    const float* x     = (const float*)d_in[0];
    const int*   ei    = (const int*)d_in[1];          // [2, NE]
    const int*   batch = (const int*)d_in[2];
    const float* Wl1 = (const float*)d_in[3];
    const float* bl1 = (const float*)d_in[4];
    const float* Wr1 = (const float*)d_in[5];
    const float* br1 = (const float*)d_in[6];
    const float* att1 = (const float*)d_in[7];
    const float* bias1 = (const float*)d_in[8];
    const float* Wl2 = (const float*)d_in[9];
    const float* bl2 = (const float*)d_in[10];
    const float* Wr2 = (const float*)d_in[11];
    const float* br2 = (const float*)d_in[12];
    const float* att2 = (const float*)d_in[13];
    const float* bias2 = (const float*)d_in[14];
    const int* src = ei;
    const int* dst = ei + NE;
    float* dout = (float*)d_out;

    // workspace carve-up (256B aligned)
    char* base = (char*)d_ws;
    size_t off = 0;
    auto carve = [&](size_t bytes) -> char* {
        char* p = base + off;
        off += (bytes + 255) & ~(size_t)255;
        return p;
    };
    _Float16* Xh      = (_Float16*)carve((size_t)NN * KPAD * 2);
    _Float16* Wpack   = (_Float16*)carve((size_t)NTILES * 3 * 512 * 2);
    float*    biascat = (float*)carve((size_t)NCAT * 4);
    float*    XLR     = (float*)carve((size_t)NN * NCAT * 4);
    float*    alpha1  = (float*)carve((size_t)NE * HEADS * 4);   // logits then alpha
    unsigned* maxenc1 = (unsigned*)carve((size_t)NN * HEADS * 4);
    float*    denom1  = (float*)carve((size_t)NN * HEADS * 4);
    float*    out1    = (float*)carve((size_t)NN * H1 * 4);
    float*    hl2     = (float*)carve((size_t)NN * 4);
    float*    hr2     = (float*)carve((size_t)NN * 4);
    float*    alpha2  = (float*)carve((size_t)NE * 4);
    unsigned* max2    = (unsigned*)carve((size_t)NN * 4);
    float*    denom2  = (float*)carve((size_t)NN * 4);
    float*    out2    = (float*)carve((size_t)NN * 4);

    const int B = 256;
    // zero accumulators + output (deterministic every call)
    zero32<<<1024, B, 0, stream>>>((unsigned*)maxenc1, (long long)NN * HEADS * 2); // maxenc1+denom1 contiguous in carve order? keep explicit:
    zero32<<<1024, B, 0, stream>>>((unsigned*)denom1, (long long)NN * HEADS);
    zero32<<<4096, B, 0, stream>>>((unsigned*)out1, (long long)NN * H1);
    zero32<<<256, B, 0, stream>>>((unsigned*)max2, NN);
    zero32<<<256, B, 0, stream>>>((unsigned*)denom2, NN);
    zero32<<<256, B, 0, stream>>>((unsigned*)out2, NN);
    zero32<<<1, B, 0, stream>>>((unsigned*)dout, NG);

    // pack inputs for WMMA
    pack_x<<<(NN * KPAD + B - 1) / B, B, 0, stream>>>(x, Xh);
    pack_w<<<(NTILES * 3 * 512 + B - 1) / B, B, 0, stream>>>(Wl1, Wr1, bl1, br1, Wpack, biascat);

    // layer-1 node transform: one fused WMMA GEMM (N = W_l || W_r)
    dim3 ggrid((MTILES + 7) / 8, NTILES);
    gemm_wmma<<<ggrid, B, 0, stream>>>(Xh, Wpack, biascat, XLR);

    // layer-1 attention + aggregation
    edge_logits1<<<(NE * HEADS + B - 1) / B, B, 0, stream>>>(src, dst, XLR, att1, alpha1, maxenc1);
    edge_alpha1<<<(NE * HEADS + B - 1) / B, B, 0, stream>>>(dst, alpha1, maxenc1, denom1);
    long long msg1 = (long long)NE * H1;
    edge_message1<<<(unsigned)((msg1 + B - 1) / B), B, 0, stream>>>(src, dst, XLR, alpha1, denom1, out1);
    long long nel = (long long)NN * H1;
    bias_elu1<<<(unsigned)((nel + B - 1) / B), B, 0, stream>>>(out1, bias1);

    // layer-2
    gemv2<<<(NN + 7) / 8, B, 0, stream>>>(out1, Wl2, bl2, Wr2, br2, hl2, hr2);
    edge_logits2<<<(NE + B - 1) / B, B, 0, stream>>>(src, dst, hl2, hr2, att2, alpha2, max2);
    edge_alpha2<<<(NE + B - 1) / B, B, 0, stream>>>(dst, alpha2, max2, denom2);
    edge_message2<<<(NE + B - 1) / B, B, 0, stream>>>(src, dst, hl2, alpha2, denom2, out2);

    // global add pool
    pool<<<(NN + B - 1) / B, B, 0, stream>>>(out2, bias2, batch, dout);
    (void)in_sizes; (void)n_in; (void)out_size; (void)ws_size;
}
